// IterativeAlignmentUnit_6579889897613
// MI455X (gfx1250) — matrix operand order repaired and checked by reading, not compiled, not run on hardware
//
#include <hip/hip_runtime.h>
#include <math.h>

// Problem constants (match reference)
#define BB   2
#define CC   64
#define HH   128
#define WW   240
#define HWW  (HH*WW)        // 30720
#define MM   (BB*HWW)       // 61440
#define OFFD 144
#define MSKD 72
#define OMD  216
#define K1   1152           // 2C*9
#define K2   576            // C*9
#define N2P  224            // conv2 N padded to 14*16

typedef __attribute__((ext_vector_type(16))) __bf16 v16bf;
typedef __attribute__((ext_vector_type(8)))  float  v8f;

// A-fragment (16x32 bf16, ISA table): elems 0..7 at base[0..7], 8..15 at base[16..23]
__device__ __forceinline__ v16bf load_a_frag(const __bf16* base) {
  union { v16bf v; uint4 q[2]; } u;
  u.q[0] = *(const uint4*)(base);
  u.q[1] = *(const uint4*)(base + 16);
  return u.v;
}
// B-fragment (32x16 bf16, ISA table): elems 0..7 at base[0..7], 8..15 at base[16..23]
__device__ __forceinline__ v16bf load_b_frag(const __bf16* base) {
  union { v16bf v; uint4 q[2]; } u;
  u.q[0] = *(const uint4*)(base);
  u.q[1] = *(const uint4*)(base + 16);
  return u.v;
}

// Raw LDS byte offset of a __shared__ object: low 32 bits of its flat address
__device__ __forceinline__ unsigned lds_off(const void* p) {
  return (unsigned)(unsigned long long)p;
}
// Async DMA: 16B from global -> LDS, tracked by ASYNCcnt (no VGPR data path)
__device__ __forceinline__ void async_b128(unsigned lds, const void* gsrc) {
  asm volatile("global_load_async_to_lds_b128 %0, %1, off"
               :: "v"(lds), "v"(gsrc) : "memory");
}
__device__ __forceinline__ void wait_async0() {
#if __has_builtin(__builtin_amdgcn_s_wait_asynccnt)
  __builtin_amdgcn_s_wait_asynccnt(0);
#else
  asm volatile("s_wait_asynccnt 0x0" ::: "memory");
#endif
}

__device__ __forceinline__ float lrelu(float v) { return v >= 0.f ? v : 0.1f*v; }

// ---------------- Kernel 0: weights fp32 -> bf16 (pad w2 to 224 rows) ----------------
__global__ __launch_bounds__(256)
void prep_weights(const float* __restrict__ w1, const float* __restrict__ w2,
                  const float* __restrict__ wd,
                  __bf16* __restrict__ w1b, __bf16* __restrict__ w2b,
                  __bf16* __restrict__ wdb) {
  int i = blockIdx.x*256 + threadIdx.x;
  if (i < 64*K1)  w1b[i] = (__bf16)w1[i];
  if (i < N2P*K2) w2b[i] = (i < OMD*K2) ? (__bf16)w2[i] : (__bf16)0.f;
  if (i < 64*K2)  wdb[i] = (__bf16)wd[i];
}

// ---------------- Kernel 1: conv1 (2C->C) + LeakyReLU, h -> bf16 ----------------
__global__ __launch_bounds__(256)
void conv1_lrelu_wmma(const float* __restrict__ nbr, const float* __restrict__ ref,
                      const __bf16* __restrict__ w1b, const float* __restrict__ b1,
                      __bf16* __restrict__ hout) {
  __shared__ __align__(16) __bf16 As[128*32];   // [row][k]
  __shared__ __align__(16) __bf16 Bs[64*32];    // [n][k]
  const int tid = threadIdx.x;
  const int lane = tid & 31, wave = tid >> 5;
  const int laneLo = lane & 15, laneHi = lane >> 4;
  const int m0 = blockIdx.x * 128;
  const int b_blk = m0 / HWW;                   // 128 | 30720: block stays in one batch
  const int rem0 = m0 - b_blk*HWW;
  const float* nbr_b = nbr + (size_t)b_blk*CC*HWW;
  const float* ref_b = ref + (size_t)b_blk*CC*HWW;

  int py[16], px[16];                            // per-i pixel coords (K-invariant)
#pragma unroll
  for (int i = 0; i < 16; ++i) {
    int rem = rem0 + wave + 8*i;
    py[i] = rem / WW; px[i] = rem - py[i]*WW;
  }
  const unsigned asB = lds_off(&Bs[0]) + tid*16; // 256 x 16B = whole B tile
  const __bf16* wrow = w1b + (size_t)(tid>>2)*K1 + (tid&3)*8;
  __bf16* asrow = &As[wave*32 + lane];           // kk = lane is K-invariant

  v8f acc[4] = {};
  for (int kb = 0; kb < K1; kb += 32) {
    async_b128(asB, wrow + kb);                  // DMA weight tile (ASYNCcnt)
    const int k = kb + lane;
    const int cin = k / 9, kid = k - cin*9;
    const int ky = kid/3 - 1, kx = kid - (kid/3)*3 - 1;
    const float* src = ((cin < CC) ? nbr_b : ref_b) + (size_t)(cin & (CC-1))*HWW;
    // Phase 1: batch all 16 gather loads (no uses -> one load clause)
    float v[16];
#pragma unroll
    for (int i = 0; i < 16; ++i) {
      int yc = min(max(py[i] + ky, 0), HH-1);
      int xc = min(max(px[i] + kx, 0), WW-1);
      v[i] = src[yc*WW + xc];
    }
    // Phase 2: validity select + cvt + LDS store (predicates recomputed, cheap)
#pragma unroll
    for (int i = 0; i < 16; ++i) {
      bool ok = ((unsigned)(py[i] + ky) < (unsigned)HH) &
                ((unsigned)(px[i] + kx) < (unsigned)WW);
      asrow[i*256] = (__bf16)(ok ? v[i] : 0.f);
    }
    wait_async0();
    __syncthreads();
    v16bf a = load_a_frag(&As[(wave*16 + laneLo)*32 + laneHi*8]);
#pragma unroll
    for (int t = 0; t < 4; ++t) {
      v16bf bf = load_b_frag(&Bs[(t*16 + laneLo)*32 + laneHi*8]);
      acc[t] = __builtin_amdgcn_wmma_f32_16x16x32_bf16(false, a, false, bf,
                                                       (short)0, acc[t], false, false);
    }
    __syncthreads();
  }
  // Epilogue: bias + leaky relu, vectorized 16B bf16 stores
#pragma unroll
  for (int t = 0; t < 4; ++t) {
    int n = t*16 + laneLo;
    float bias = b1[n];
    __bf16* dst = hout + (size_t)(b_blk*CC + n)*HWW + rem0 + wave*16 + 8*laneHi;
    union { __bf16 h[8]; uint4 q; } o;
#pragma unroll
    for (int r = 0; r < 8; ++r) o.h[r] = (__bf16)lrelu(acc[t][r] + bias);
    *(uint4*)dst = o.q;
  }
}

// ---------------- Kernel 2: conv2 (C->216): offsets fp32 -> d_out, sigmoid masks -> ws ----------------
__global__ __launch_bounds__(256)
void conv2_offsetmask_wmma(const __bf16* __restrict__ hin,
                           const __bf16* __restrict__ w2b, const float* __restrict__ b2,
                           float* __restrict__ off_out,   // [B,144,H,W]
                           float* __restrict__ mask_ws) { // [B,72,H,W]
  __shared__ __align__(16) __bf16 As[64*32];
  __shared__ __align__(16) __bf16 Bs[N2P*32];
  const int tid = threadIdx.x;
  const int lane = tid & 31, wave = tid >> 5;
  const int laneLo = lane & 15, laneHi = lane >> 4;
  const int mTile = wave & 3, nHalf = wave >> 2;
  const int m0 = blockIdx.x * 64;
  const int b_blk = m0 / HWW;
  const int rem0 = m0 - b_blk*HWW;
  const __bf16* hin_b = hin + (size_t)b_blk*CC*HWW;

  int py[8], px[8];
#pragma unroll
  for (int i = 0; i < 8; ++i) {
    int rem = rem0 + wave + 8*i;
    py[i] = rem / WW; px[i] = rem - py[i]*WW;
  }
  const unsigned ldsB = lds_off(&Bs[0]);
  __bf16* asrow = &As[wave*32 + lane];

  v8f acc[7] = {};
  for (int kb = 0; kb < K2; kb += 32) {
    // DMA B tile: 224 rows x 64B = 896 x 16B chunks
#pragma unroll
    for (int j = 0; j < 4; ++j) {
      int c = tid + j*256;
      if (c < (N2P*32*2)/16)
        async_b128(ldsB + c*16, w2b + (size_t)(c>>2)*K2 + kb + (c&3)*8);
    }
    const int k = kb + lane;
    const int cin = k / 9, kid = k - cin*9;
    const int ky = kid/3 - 1, kx = kid - (kid/3)*3 - 1;
    const __bf16* src = hin_b + (size_t)cin*HWW;
    float v[8];
#pragma unroll
    for (int i = 0; i < 8; ++i) {
      int yc = min(max(py[i] + ky, 0), HH-1);
      int xc = min(max(px[i] + kx, 0), WW-1);
      v[i] = (float)src[yc*WW + xc];
    }
#pragma unroll
    for (int i = 0; i < 8; ++i) {
      bool ok = ((unsigned)(py[i] + ky) < (unsigned)HH) &
                ((unsigned)(px[i] + kx) < (unsigned)WW);
      asrow[i*256] = (__bf16)(ok ? v[i] : 0.f);
    }
    wait_async0();
    __syncthreads();
    v16bf a = load_a_frag(&As[(mTile*16 + laneLo)*32 + laneHi*8]);
#pragma unroll
    for (int t = 0; t < 7; ++t) {
      int nt = nHalf*7 + t;
      v16bf bf = load_b_frag(&Bs[(nt*16 + laneLo)*32 + laneHi*8]);
      acc[t] = __builtin_amdgcn_wmma_f32_16x16x32_bf16(false, a, false, bf,
                                                       (short)0, acc[t], false, false);
    }
    __syncthreads();
  }
#pragma unroll
  for (int t = 0; t < 7; ++t) {
    int n = (nHalf*7 + t)*16 + laneLo;
    if (n < OMD) {
      float bias = b2[n];
      int rem = rem0 + mTile*16 + 8*laneHi;
      union { float f[8]; float4 q[2]; } o;
      if (n < OFFD) {
#pragma unroll
        for (int r = 0; r < 8; ++r) o.f[r] = acc[t][r] + bias;
        float4* dst = (float4*)(off_out + (size_t)(b_blk*OFFD + n)*HWW + rem);
        dst[0] = o.q[0]; dst[1] = o.q[1];
      } else {
#pragma unroll
        for (int r = 0; r < 8; ++r) o.f[r] = 1.f / (1.f + __expf(-(acc[t][r] + bias)));
        float4* dst = (float4*)(mask_ws + (size_t)(b_blk*MSKD + (n - OFFD))*HWW + rem);
        dst[0] = o.q[0]; dst[1] = o.q[1];
      }
    }
  }
}

// ---------------- Kernel 3: modulated deformable conv (C->C) ----------------
__global__ __launch_bounds__(256)
void dconv_wmma(const float* __restrict__ nbr,
                const float* __restrict__ off,    // [B,144,H,W]
                const float* __restrict__ mask,   // [B,72,H,W]
                const __bf16* __restrict__ wdb, const float* __restrict__ bd,
                float* __restrict__ out) {        // [B,64,H,W]
  __shared__ __align__(16) __bf16 As[128*32];
  __shared__ __align__(16) __bf16 Bs[64*32];
  const int tid = threadIdx.x;
  const int lane = tid & 31, wave = tid >> 5;
  const int laneLo = lane & 15, laneHi = lane >> 4;
  const int m0 = blockIdx.x * 128;
  const int b_blk = m0 / HWW;
  const int rem0 = m0 - b_blk*HWW;
  const float* nbr_b  = nbr  + (size_t)b_blk*CC*HWW;
  const float* off_b  = off  + (size_t)b_blk*OFFD*HWW;
  const float* mask_b = mask + (size_t)b_blk*MSKD*HWW;

  int py[16], px[16];
#pragma unroll
  for (int i = 0; i < 16; ++i) {
    int rem = rem0 + wave + 8*i;
    py[i] = rem / WW; px[i] = rem - py[i]*WW;
  }
  const unsigned asB = lds_off(&Bs[0]) + tid*16;
  const __bf16* wrow = wdb + (size_t)(tid>>2)*K2 + (tid&3)*8;
  __bf16* asrow = &As[wave*32 + lane];

  v8f acc[4] = {};
  for (int kb = 0; kb < K2; kb += 32) {
    async_b128(asB, wrow + kb);
    const int k = kb + lane;
    const int cin = k / 9, kid = k - cin*9;
    const int g = cin >> 3;
    const int ky = kid/3 - 1, kx = kid - (kid/3)*3 - 1;
    const float* srcC = nbr_b + (size_t)cin*HWW;
    const float* offY = off_b + (size_t)(g*18 + kid)*HWW;
    const float* offX = offY + (size_t)9*HWW;
    const float* mskC = mask_b + (size_t)(g*9 + kid)*HWW;

    // Phase 1: batch all offset/mask loads (48 loads, one clause)
    float dy[16], dx[16], mv[16];
#pragma unroll
    for (int i = 0; i < 16; ++i) {
      int rem = rem0 + wave + 8*i;
      dy[i] = offY[rem]; dx[i] = offX[rem]; mv[i] = mskC[rem];
    }
    // Phase 2/3 per group of 4 pixels: batch 16 corner loads, then combine
#pragma unroll
    for (int ii = 0; ii < 16; ii += 4) {
      float c00[4], c01[4], c10[4], c11[4], wy[4], wx[4];
      int y0a[4], x0a[4];
#pragma unroll
      for (int j = 0; j < 4; ++j) {
        int i = ii + j;
        float pyf = dy[i] + (float)(py[i] + ky);
        float pxf = dx[i] + (float)(px[i] + kx);
        float y0f = floorf(pyf), x0f = floorf(pxf);
        wy[j] = pyf - y0f; wx[j] = pxf - x0f;
        int y0 = (int)y0f, x0 = (int)x0f;
        y0a[j] = y0; x0a[j] = x0;
        int y0c = min(max(y0, 0), HH-1),   x0c = min(max(x0, 0), WW-1);
        int y1c = min(max(y0+1, 0), HH-1), x1c = min(max(x0+1, 0), WW-1);
        c00[j] = srcC[y0c*WW + x0c];
        c01[j] = srcC[y0c*WW + x1c];
        c10[j] = srcC[y1c*WW + x0c];
        c11[j] = srcC[y1c*WW + x1c];
      }
#pragma unroll
      for (int j = 0; j < 4; ++j) {
        int y0 = y0a[j], x0 = x0a[j];
        bool y0v = (unsigned)y0 < (unsigned)HH, y1v = (unsigned)(y0+1) < (unsigned)HH;
        bool x0v = (unsigned)x0 < (unsigned)WW, x1v = (unsigned)(x0+1) < (unsigned)WW;
        float v00 = (y0v && x0v) ? c00[j] : 0.f;
        float v01 = (y0v && x1v) ? c01[j] : 0.f;
        float v10 = (y1v && x0v) ? c10[j] : 0.f;
        float v11 = (y1v && x1v) ? c11[j] : 0.f;
        float s = (v00*(1.f-wy[j])*(1.f-wx[j]) + v01*(1.f-wy[j])*wx[j]
                 + v10*wy[j]*(1.f-wx[j])       + v11*wy[j]*wx[j]) * mv[ii+j];
        asrow[(ii+j)*256] = (__bf16)s;
      }
    }
    wait_async0();
    __syncthreads();
    v16bf a = load_a_frag(&As[(wave*16 + laneLo)*32 + laneHi*8]);
#pragma unroll
    for (int t = 0; t < 4; ++t) {
      v16bf bf = load_b_frag(&Bs[(t*16 + laneLo)*32 + laneHi*8]);
      acc[t] = __builtin_amdgcn_wmma_f32_16x16x32_bf16(false, a, false, bf,
                                                       (short)0, acc[t], false, false);
    }
    __syncthreads();
  }
#pragma unroll
  for (int t = 0; t < 4; ++t) {
    int n = t*16 + laneLo;
    float bias = bd[n];
    union { float f[8]; float4 q[2]; } o;
#pragma unroll
    for (int r = 0; r < 8; ++r) o.f[r] = acc[t][r] + bias;
    float4* dst = (float4*)(out + (size_t)(b_blk*CC + n)*HWW + rem0 + wave*16 + 8*laneHi);
    dst[0] = o.q[0]; dst[1] = o.q[1];
  }
}

extern "C" void kernel_launch(void* const* d_in, const int* in_sizes, int n_in,
                              void* d_out, int out_size, void* d_ws, size_t ws_size,
                              hipStream_t stream) {
  const float* nbr = (const float*)d_in[0];
  const float* ref = (const float*)d_in[1];
  const float* w1  = (const float*)d_in[2];
  const float* b1  = (const float*)d_in[3];
  const float* w2  = (const float*)d_in[4];
  const float* b2  = (const float*)d_in[5];
  const float* wd  = (const float*)d_in[6];
  const float* bd  = (const float*)d_in[7];

  float* aligned = (float*)d_out;                      // [B,64,H,W]
  float* off_out = aligned + (size_t)BB*CC*HWW;        // [B,144,H,W]

  char* ws = (char*)d_ws;
  size_t o = 0;
  __bf16* h_ws    = (__bf16*)(ws + o); o += (size_t)BB*CC*HWW*2;    // 7.86 MB
  float*  mask_ws = (float*) (ws + o); o += (size_t)BB*MSKD*HWW*4;  // 17.69 MB
  __bf16* w1b     = (__bf16*)(ws + o); o += (size_t)64*K1*2;
  __bf16* w2b     = (__bf16*)(ws + o); o += (size_t)N2P*K2*2;
  __bf16* wdb     = (__bf16*)(ws + o);

  prep_weights         <<<(N2P*K2 + 255)/256, 256, 0, stream>>>(w1, w2, wd, w1b, w2b, wdb);
  conv1_lrelu_wmma     <<<MM/128, 256, 0, stream>>>(nbr, ref, w1b, b1, h_ws);
  conv2_offsetmask_wmma<<<MM/64,  256, 0, stream>>>(h_ws, w2b, b2, off_out, mask_ws);
  dconv_wmma           <<<MM/128, 256, 0, stream>>>(nbr, off_out, mask_ws, wdb, bd, aligned);
}
